// CausalBiBCNAttention_82360292868513
// MI455X (gfx1250) — matrix operand order repaired and checked by reading, not compiled, not run on hardware
//
#include <hip/hip_runtime.h>
#include <hip/hip_bf16.h>

typedef __bf16 bf16;
typedef __attribute__((ext_vector_type(16))) __bf16 v16bf;
typedef __attribute__((ext_vector_type(8)))  __bf16 v8bf;
typedef __attribute__((ext_vector_type(4)))  __bf16 v4bf;
typedef __attribute__((ext_vector_type(8)))  float  v8f;

// ---- CDNA5 async global->LDS copy (ASYNCcnt-tracked), inline asm ----
__device__ __forceinline__ void async_ld_b128(const bf16* g, unsigned lds_off) {
  asm volatile("global_load_async_to_lds_b128 %0, %1, off"
               :: "v"(lds_off), "v"(g) : "memory");
}
template<int N>
__device__ __forceinline__ void wait_asynccnt() {
  asm volatile("s_wait_asynccnt %0" :: "i"(N) : "memory");
}

// ---------------- fp32 -> bf16 conversion (vectorized x4) ----------------
__global__ __launch_bounds__(256)
void cvt_f32_bf16(const float* __restrict__ in, bf16* __restrict__ out, long n4) {
  long i = (long)blockIdx.x * blockDim.x + threadIdx.x;
  if (i >= n4) return;
  float4 v = ((const float4*)in)[i];
  v4bf o;
  o.x = (bf16)v.x; o.y = (bf16)v.y; o.z = (bf16)v.z; o.w = (bf16)v.w;
  ((v4bf*)out)[i] = o;
}

// ---------------- fp32 [D,R] -> bf16 [R,D] transpose+convert ----------------
__global__ __launch_bounds__(256)
void cvt_transpose_bf16(const float* __restrict__ in, bf16* __restrict__ out,
                        int D_, int R_) {
  __shared__ float tile[32][33];
  const int r0 = blockIdx.x * 32;     // R dim
  const int d0 = blockIdx.y * 32;     // D dim
  const int tx = threadIdx.x & 31, ty = threadIdx.x >> 5;
  for (int i = ty; i < 32; i += 8)
    tile[i][tx] = in[(long)(d0 + i) * R_ + r0 + tx];
  __syncthreads();
  for (int i = ty; i < 32; i += 8)
    out[(long)(r0 + i) * D_ + d0 + tx] = (bf16)tile[tx][i];
}

// ---------------- chunked causal scan (3 passes) ----------------
#define SCH 128   // rows per chunk

__global__ __launch_bounds__(256)
void scan_partials(const float* __restrict__ kW, const float* __restrict__ ikV,
                   const int* __restrict__ mask,
                   float* __restrict__ pK, float* __restrict__ pIK, int T, int R) {
  const int b = blockIdx.y, c = blockIdx.x, r = threadIdx.x;
  float sk = 0.f, sik = 0.f;
  for (int t = c * SCH; t < (c + 1) * SCH; ++t) {
    const long row = (long)b * T + t;
    const float m = (float)mask[row];
    const long idx = row * R + r;
    sk  += kW[idx]  * m;
    sik += ikV[idx] * m;
  }
  const long o = ((long)b * gridDim.x + c) * R + r;
  pK[o] = sk; pIK[o] = sik;
}

__global__ __launch_bounds__(256)
void scan_offsets(float* __restrict__ pK, float* __restrict__ pIK, int nc, int R) {
  const int b = blockIdx.x, r = threadIdx.x;
  float ak = 0.f, aik = 0.f;
  for (int c = 0; c < nc; ++c) {
    const long o = ((long)b * nc + c) * R + r;
    const float k = pK[o], ik = pIK[o];
    pK[o] = ak; pIK[o] = aik;
    ak += k; aik += ik;
  }
}

__global__ void mask_scan(const int* __restrict__ mask, float* __restrict__ nvec,
                          int Bb, int T) {
  const int b = threadIdx.x;
  if (b >= Bb) return;
  float c = 0.f;
  for (int t = 0; t < T; ++t) {
    const long row = (long)b * T + t;
    c += (float)mask[row];
    nvec[row] = c;
  }
}

__global__ __launch_bounds__(256)
void scan_final(const float* __restrict__ qV, const float* __restrict__ kW,
                const float* __restrict__ ikV, const float* __restrict__ iqW,
                const int* __restrict__ mask, const float* __restrict__ alphap,
                const float* __restrict__ pK, const float* __restrict__ pIK,
                bf16* __restrict__ P, int T, int R) {
  const int b = blockIdx.y, c = blockIdx.x, r = threadIdx.x;
  const float alpha = *alphap;
  const long o = ((long)b * gridDim.x + c) * R + r;
  float ck = pK[o], cik = pIK[o];
  for (int t = c * SCH; t < (c + 1) * SCH; ++t) {
    const long row = (long)b * T + t;
    const long idx = row * R + r;
    const float m = (float)mask[row];
    ck  += kW[idx]  * m;
    cik += ikV[idx] * m;
    P[idx] = (bf16)(qV[idx] * ck + alpha * cik * iqW[idx]);
  }
}

// ---------------- tiled bf16 WMMA GEMM, double-buffered async LDS ----------------
// C[M,N] = A[M,K] @ B^T, B is [N,K] row-major.
// OUTMODE 0: fp32 out   1: bf16 out   2: bf16 out with (acc + n*(1+a)*bias)/max(n,1)
template<int OUTMODE>
__global__ __launch_bounds__(256)
void gemm_bf16_wmma(const bf16* __restrict__ A, const bf16* __restrict__ B,
                    float* __restrict__ Cf, bf16* __restrict__ Cb,
                    int M, int N, int K, int ldb,
                    const float* __restrict__ nvec, const float* __restrict__ bias,
                    const float* __restrict__ alphap) {
  constexpr int BM = 128, BN = 128, BK = 64, LDT = 72;  // 144B pitch, 16B aligned
  constexpr int TILE = BM * LDT;                        // bf16 elems per buffer
  constexpr int NASYNC = 8;                             // async ops per stage per thread
  __shared__ bf16 As[2 * TILE];
  __shared__ bf16 Bs[2 * TILE];

  const int tid  = threadIdx.x;
  const int lane = tid & 31;
  const int wave = tid >> 5;
  const int wm   = wave & 3;        // 4 waves along M -> 32 rows each
  const int wn   = wave >> 2;       // 2 waves along N -> 64 cols each
  const int r16  = lane & 15;
  const int kg   = lane >> 4;
  const long mBase = (long)blockIdx.y * BM;
  const long nBase = (long)blockIdx.x * BN;

  // stage one (BMxBK, BNxBK) tile pair into buffer `buf`, all-async
  auto stage = [&](int buf, int k0) {
    #pragma unroll
    for (int s = 0; s < 4; ++s) {                  // 1024 16B-chunks / 256 thr
      const int c = tid + s * 256;
      const int row = c >> 3, col8 = (c & 7) * 8;
      async_ld_b128(A + (mBase + row) * (long)K + k0 + col8,
          (unsigned)(unsigned long long)&As[buf * TILE + row * LDT + col8]);
    }
    #pragma unroll
    for (int s = 0; s < 4; ++s) {
      const int c = tid + s * 256;
      const int row = c >> 3, col8 = (c & 7) * 8;
      async_ld_b128(B + (nBase + row) * (long)ldb + k0 + col8,
          (unsigned)(unsigned long long)&Bs[buf * TILE + row * LDT + col8]);
    }
  };

  v8f acc[2][4];
  const v8f vz = {0.f,0.f,0.f,0.f,0.f,0.f,0.f,0.f};
  for (int i = 0; i < 2; ++i)
    for (int j = 0; j < 4; ++j) acc[i][j] = vz;

  auto compute = [&](int buf) {
    #pragma unroll
    for (int kk = 0; kk < BK; kk += 32) {          // 2 WMMA k-steps per stage
      v16bf af[2], bfr[4];
      #pragma unroll
      for (int i = 0; i < 2; ++i) {
        const bf16* base = &As[buf * TILE + (wm * 32 + i * 16 + r16) * LDT + kk];
        *(v8bf*)&af[i]       = *(const v8bf*)(base + kg * 8);
        *((v8bf*)&af[i] + 1) = *(const v8bf*)(base + 16 + kg * 8);
      }
      #pragma unroll
      for (int j = 0; j < 4; ++j) {
        const bf16* base = &Bs[buf * TILE + (wn * 64 + j * 16 + r16) * LDT + kk + kg * 8];
        *(v8bf*)&bfr[j]       = *(const v8bf*)(base);
        *((v8bf*)&bfr[j] + 1) = *(const v8bf*)(base + 16);
      }
      #pragma unroll
      for (int i = 0; i < 2; ++i)
        #pragma unroll
        for (int j = 0; j < 4; ++j)
          acc[i][j] = __builtin_amdgcn_wmma_f32_16x16x32_bf16(
              false, af[i], false, bfr[j], (short)0, acc[i][j], false, false);
    }
  };

  // software pipeline: branch-free steady state, peeled epilogue
  stage(0, 0);
  int buf = 0;
  for (int k0 = 0; k0 + BK < K; k0 += BK) {
    stage(buf ^ 1, k0 + BK);       // prefetch next tile (8 async ops)
    wait_asynccnt<NASYNC>();       // my current-buf copies have landed
    __syncthreads();               // everyone's copies have landed
    compute(buf);
    __syncthreads();               // done reading buf before it is re-staged
    buf ^= 1;
  }
  wait_asynccnt<0>();
  __syncthreads();
  compute(buf);

  // store: C/D layout => VGPR r: M = r + 8*laneHi, N = lane&15
  float a1 = 0.f;
  if (OUTMODE == 2) a1 = 1.f + *alphap;
  for (int i = 0; i < 2; ++i)
    for (int j = 0; j < 4; ++j) {
      const long row0 = mBase + wm * 32 + i * 16 + kg * 8;
      const long col  = nBase + wn * 64 + j * 16 + r16;
      if constexpr (OUTMODE == 0) {
        for (int r = 0; r < 8; ++r)
          Cf[(row0 + r) * (long)N + col] = acc[i][j][r];
      } else if constexpr (OUTMODE == 1) {
        for (int r = 0; r < 8; ++r)
          Cb[(row0 + r) * (long)N + col] = (bf16)acc[i][j][r];
      } else {
        const float bc = bias[col];
        for (int r = 0; r < 8; ++r) {
          const float nv = nvec[row0 + r];
          const float y  = (acc[i][j][r] + nv * a1 * bc) / fmaxf(nv, 1.f);
          Cb[(row0 + r) * (long)N + col] = (bf16)y;
        }
      }
    }
}

// ---------------- launch ----------------
extern "C" void kernel_launch(void* const* d_in, const int* in_sizes, int n_in,
                              void* d_out, int out_size, void* d_ws, size_t ws_size,
                              hipStream_t stream) {
  const float* x     = (const float*)d_in[0];
  const float* Wq    = (const float*)d_in[1];
  const float* Wk    = (const float*)d_in[2];
  const float* Wo    = (const float*)d_in[3];
  const float* Winv  = (const float*)d_in[4];
  const float* U     = (const float*)d_in[5];
  const float* V     = (const float*)d_in[6];
  const float* Wm    = (const float*)d_in[7];
  const float* bias  = (const float*)d_in[8];
  const float* alpha = (const float*)d_in[9];
  const int*   mask  = (const int*)d_in[10];
  float* out = (float*)d_out;

  const int Bb = 4, T = 4096, D = 1024, R = 256;
  const int M = Bb * T;                     // 16384
  const int NC = T / SCH;                   // 32 chunks

  char* ws = (char*)d_ws;
  const size_t MB = 1ull << 20;
  bf16*  xbf  = (bf16*)(ws + 0);            // 32MB, dead after Q/K GEMMs
  bf16*  wqbf = (bf16*)(ws + 32 * MB);
  bf16*  wkbf = (bf16*)(ws + 34 * MB);
  bf16*  wobf = (bf16*)(ws + 36 * MB);
  bf16*  wibf = (bf16*)(ws + 38 * MB);
  bf16*  ubf  = (bf16*)(ws + 40 * MB);      // U stays [D,R] (already [N,K] for Y GEMM)
  bf16*  vtb  = (bf16*)(ws + 41 * MB);      // V^T  [R,D] bf16
  bf16*  wmtb = (bf16*)(ws + 42 * MB);      // Wm^T [R,D] bf16
  bf16*  Qbf  = (bf16*)(ws + 48 * MB);      // 32MB, dead after qV GEMM
  bf16*  Kbf  = (bf16*)(ws + 80 * MB);
  bf16*  iQbf = (bf16*)(ws + 112 * MB);
  bf16*  iKbf = (bf16*)(ws + 144 * MB);
  float* qV   = (float*)(ws + 176 * MB);    // 16MB each
  float* kW   = (float*)(ws + 192 * MB);
  float* ikV  = (float*)(ws + 208 * MB);
  float* iqW  = (float*)(ws + 224 * MB);
  bf16*  Pbf  = (bf16*)(ws + 0);            // reuse x region (8MB)
  float* nvec = (float*)(ws + 9 * MB);      // reuse x region (64KB)
  float* pK   = (float*)(ws + 10 * MB);     // 128KB partials
  float* pIK  = (float*)(ws + 10 * MB + 512 * 1024);
  bf16*  Ybf  = (bf16*)(ws + 48 * MB);      // reuse Q region (32MB)

  auto cdiv = [](long a, long b) { return (int)((a + b - 1) / b); };

  // 1) convert inputs/weights to bf16; transpose V, Wm so every GEMM is A@B^T
  cvt_f32_bf16<<<cdiv((long)M * D / 4, 256), 256, 0, stream>>>(x, xbf, (long)M * D / 4);
  cvt_f32_bf16<<<cdiv((long)D * D / 4, 256), 256, 0, stream>>>(Wq,   wqbf, (long)D * D / 4);
  cvt_f32_bf16<<<cdiv((long)D * D / 4, 256), 256, 0, stream>>>(Wk,   wkbf, (long)D * D / 4);
  cvt_f32_bf16<<<cdiv((long)D * D / 4, 256), 256, 0, stream>>>(Wo,   wobf, (long)D * D / 4);
  cvt_f32_bf16<<<cdiv((long)D * D / 4, 256), 256, 0, stream>>>(Winv, wibf, (long)D * D / 4);
  cvt_f32_bf16<<<cdiv((long)D * R / 4, 256), 256, 0, stream>>>(U, ubf, (long)D * R / 4);
  cvt_transpose_bf16<<<dim3(R / 32, D / 32), 256, 0, stream>>>(V,  vtb,  D, R);
  cvt_transpose_bf16<<<dim3(R / 32, D / 32), 256, 0, stream>>>(Wm, wmtb, D, R);

  dim3 blk(256);
  dim3 gD(D / 128, M / 128);   // N = 1024
  dim3 gR(R / 128, M / 128);   // N = 256

  // 2) Q = x @ Wq^T ; K = x @ Wk^T   (bf16 out)
  gemm_bf16_wmma<1><<<gD, blk, 0, stream>>>(xbf, wqbf, nullptr, Qbf, M, D, D, D, nullptr, nullptr, nullptr);
  gemm_bf16_wmma<1><<<gD, blk, 0, stream>>>(xbf, wkbf, nullptr, Kbf, M, D, D, D, nullptr, nullptr, nullptr);

  // 3) invQ = Q @ Winv^T ; invK = K @ Winv^T
  gemm_bf16_wmma<1><<<gD, blk, 0, stream>>>(Qbf, wibf, nullptr, iQbf, M, D, D, D, nullptr, nullptr, nullptr);
  gemm_bf16_wmma<1><<<gD, blk, 0, stream>>>(Kbf, wibf, nullptr, iKbf, M, D, D, D, nullptr, nullptr, nullptr);

  // 4) rank-R projections: A @ (X^T)^T = A @ X   (fp32 out)
  gemm_bf16_wmma<0><<<gR, blk, 0, stream>>>(Qbf,  vtb,  qV,  nullptr, M, R, D, D, nullptr, nullptr, nullptr);
  gemm_bf16_wmma<0><<<gR, blk, 0, stream>>>(Kbf,  wmtb, kW,  nullptr, M, R, D, D, nullptr, nullptr, nullptr);
  gemm_bf16_wmma<0><<<gR, blk, 0, stream>>>(iKbf, vtb,  ikV, nullptr, M, R, D, D, nullptr, nullptr, nullptr);
  gemm_bf16_wmma<0><<<gR, blk, 0, stream>>>(iQbf, wmtb, iqW, nullptr, M, R, D, D, nullptr, nullptr, nullptr);

  // 5) chunked masked causal scan + combine -> P (bf16), n
  scan_partials<<<dim3(NC, Bb), blk, 0, stream>>>(kW, ikV, mask, pK, pIK, T, R);
  scan_offsets<<<Bb, blk, 0, stream>>>(pK, pIK, NC, R);
  mask_scan<<<1, Bb, 0, stream>>>(mask, nvec, Bb, T);
  scan_final<<<dim3(NC, Bb), blk, 0, stream>>>(qV, kW, ikV, iqW, mask, alpha, pK, pIK, Pbf, T, R);

  // 6) Y = (P @ U^T + n*(1+a)*bias) / max(n,1)   (bf16 out, fused epilogue)
  gemm_bf16_wmma<2><<<gD, blk, 0, stream>>>(Pbf, ubf, nullptr, Ybf, M, D, R, R, nvec, bias, alpha);

  // 7) out = Y @ Wo^T   (fp32 out)
  gemm_bf16_wmma<0><<<gD, blk, 0, stream>>>(Ybf, wobf, out, nullptr, M, D, D, D, nullptr, nullptr, nullptr);
}
